// BooleanAnchorTable_4681514352961
// MI455X (gfx1250) — compile-verified, matrix operand order checked by reading, not run on hardware
//
#include <hip/hip_runtime.h>

// Problem geometry (fixed by the reference).
#define H      256      // hidden size (GEMM K dim)
#define KCB    8192     // codebook size (GEMM N dim)
#define NTOK   32768    // 8*4096 tokens (GEMM M dim)
#define MB     128      // token rows per workgroup (8 waves x 16 rows)

// LDS staging of the codebook (double-buffered, async DMA).
#define NB     64                  // codes staged per block
#define LROW   264                 // bf16 elements per padded LDS row (256 + 8)
#define ROWB   (LROW * 2)          // 528 bytes per row: 132 dwords -> bank-optimal
#define HALFE  (NB * LROW)         // elems in the hi half of a slot (16896)
#define SLOTE  (NB * LROW * 2)     // elems per slot (hi+lo) = 33792
#define NBLK   (KCB / NB)          // 128 staged blocks
#define LDSB   (2 * SLOTE * 2)     // dynamic LDS bytes (two slots) = 135168

typedef __bf16 bf16_t;
typedef __attribute__((ext_vector_type(16))) __bf16 v16bf;
typedef __attribute__((ext_vector_type(8)))  __bf16 v8bf;
typedef __attribute__((ext_vector_type(8)))  float  v8f;

union V16 { v16bf v; v8bf h[2]; };

// Map K index -> position inside the 16-bit WMMA A/B fragment stream.
// Fragment (chunk c, lane-half hf) holds K = c*32 + hf*8 + j (slots 0..7)
// and K = c*32 + 16 + hf*8 + j (slots 8..15).
__device__ __forceinline__ int ileave(int k) {
  int c = k >> 5, r = k & 31;
  int hf, slot;
  if (r < 16) { hf = r >> 3;        slot = r & 7; }
  else        { hf = (r - 16) >> 3; slot = 8 + ((r - 16) & 7); }
  return c * 32 + hf * 16 + slot;
}

__device__ __forceinline__ v16bf load16(const bf16_t* p) {
  V16 u;
  u.h[0] = *reinterpret_cast<const v8bf*>(p);
  u.h[1] = *reinterpret_cast<const v8bf*>(p + 8);
  return u.v;
}

// One 16B/lane async DMA transfer: global (saddr + voff) -> LDS (ldsoff).
// Tracked by ASYNCcnt; no VGPR data round-trip.
__device__ __forceinline__ void async16(uint32_t ldsoff, uint32_t goff, const void* sbase) {
  asm volatile("global_load_async_to_lds_b128 %0, %1, %2"
               :: "v"(ldsoff), "v"(goff), "s"(sbase)
               : "memory");
}

__device__ __forceinline__ void wait_async0() {
  asm volatile("s_wait_asynccnt 0x0" ::: "memory");
}

// Stage NB codes (hi+lo, 64 KiB) into one LDS slot. Global data is linear
// (512 B per code); LDS rows are padded to 528 B, so the LDS offset carries
// the per-row pad while the global offset stays linear. Row index per
// (wave, i) is lane-uniform: row = wave*8 + i.
__device__ __forceinline__ void stage_block(uint32_t slot_byte,
                                            const bf16_t* __restrict__ ghi,
                                            const bf16_t* __restrict__ glo,
                                            int blk, int wave, int lane) {
  const uint32_t gv = (uint32_t)blk * (NB * H * 2) + (uint32_t)(wave * 256 + lane) * 16u;
  const uint32_t lh = slot_byte + (uint32_t)(wave * 8) * ROWB + (uint32_t)lane * 16u;
  const uint32_t ll = lh + (uint32_t)HALFE * 2u;
#pragma unroll
  for (int i = 0; i < 8; ++i) async16(lh + (uint32_t)i * ROWB, gv + (uint32_t)i * 512u, ghi);
#pragma unroll
  for (int i = 0; i < 8; ++i) async16(ll + (uint32_t)i * ROWB, gv + (uint32_t)i * 512u, glo);
}

// ---------------------------------------------------------------------------
// Pass 1: emb -> bf16 hi/lo (fragment-interleaved) + per-code squared norms.
// ---------------------------------------------------------------------------
__global__ __launch_bounds__(256) void vq_prep(const float* __restrict__ emb,
                                               bf16_t* __restrict__ ehi,
                                               bf16_t* __restrict__ elo,
                                               float* __restrict__ enorm) {
  const int code = blockIdx.x, k = threadIdx.x;
  float x = emb[(size_t)code * H + k];
  bf16_t hb = (bf16_t)x;
  float  hf = (float)hb;
  bf16_t lb = (bf16_t)(x - hf);
  int p = code * H + ileave(k);
  ehi[p] = hb;
  elo[p] = lb;
  __shared__ float red[256];
  red[k] = x * x;
  __syncthreads();
  for (int s = 128; s > 0; s >>= 1) {
    if (k < s) red[k] += red[k + s];
    __syncthreads();
  }
  if (k == 0) enorm[code] = red[0];
}

// ---------------------------------------------------------------------------
// Pass 2: fused GEMM + argmin.  A (hi+lo) is built once into 128 VGPRs per
// wave; the codebook streams through double-buffered LDS slots filled by
// GLOBAL_LOAD_ASYNC_TO_LDS_B128 (one copy per workgroup instead of one per
// wave), overlapped with 3x bf16 WMMAs per K=32 chunk (split-precision
// hi*hi + lo*hi + hi*lo ~= f32 dot).  The chunk loop is software-pipelined
// one deep so ds_loads for chunk c+1 issue before the wait on chunk c.
// ---------------------------------------------------------------------------
__global__ __launch_bounds__(256) void vq_argmin(const float* __restrict__ z,
                                                 const bf16_t* __restrict__ ehi,
                                                 const bf16_t* __restrict__ elo,
                                                 const float* __restrict__ enorm,
                                                 int* __restrict__ didx) {
  extern __shared__ bf16_t smem[];
  // Truncating the flat shared address to 32 bits yields the wave-relative
  // LDS byte offset (flat-LDS addressing uses addr[31:0]).
  const uint32_t smem_byte = (uint32_t)(uintptr_t)smem;

  const int tid  = threadIdx.x;
  const int wave = tid >> 5;
  const int lane = tid & 31;
  const int m    = lane & 15;   // A-fragment row within the 16-row tile
  const int hf   = lane >> 4;   // lane half selects K sub-pattern
  const int arow = blockIdx.x * MB + wave * 16 + m;
  const float* zrow = z + (size_t)arow * H;

  // Build A fragments (hi and lo) for all 8 K-chunks, held in registers.
  v16bf Ahi[8], Alo[8];
#pragma unroll
  for (int c = 0; c < 8; ++c) {
    V16 uh, ul;
#pragma unroll
    for (int j = 0; j < 8; ++j) {
      float x  = zrow[c * 32 + hf * 8 + j];
      bf16_t a = (bf16_t)x;
      uh.v[j]     = a;
      ul.v[j]     = (bf16_t)(x - (float)a);
      float y  = zrow[c * 32 + 16 + hf * 8 + j];
      bf16_t b = (bf16_t)y;
      uh.v[8 + j] = b;
      ul.v[8 + j] = (bf16_t)(y - (float)b);
    }
    Ahi[c] = uh.v;
    Alo[c] = ul.v;
  }

  float best[8];
  int   bidx[8];
#pragma unroll
  for (int r = 0; r < 8; ++r) { best[r] = 3.4e38f; bidx[r] = 0; }

  const int n = lane & 15;  // output column (code within tile) for this lane

  // Prologue: stage block 0 into slot 0.
  stage_block(smem_byte, ehi, elo, 0, wave, lane);
  wait_async0();
  __syncthreads();

  for (int blk = 0; blk < NBLK; ++blk) {
    const int buf = blk & 1;
    if (blk + 1 < NBLK)
      stage_block(smem_byte + (uint32_t)(buf ^ 1) * (SLOTE * 2), ehi, elo,
                  blk + 1, wave, lane);

    const bf16_t* sbase = smem + (size_t)buf * SLOTE;
#pragma unroll
    for (int t = 0; t < NB / 16; ++t) {
      const int rloc = t * 16 + n;
      const int code = blk * NB + rloc;
      const bf16_t* bh = sbase + rloc * LROW + hf * 16;
      const bf16_t* bl = bh + HALFE;
      v8f acc = {};
      // 1-deep software pipeline over the 8 K-chunks: two alternating
      // B-fragment register sets so next loads issue before current wait.
      v16bf Bhi0 = load16(bh);
      v16bf Blo0 = load16(bl);
#pragma unroll
      for (int c = 0; c < 8; ++c) {
        v16bf Bhi1 = Bhi0, Blo1 = Blo0;
        if (c + 1 < 8) {
          Bhi1 = load16(bh + (c + 1) * 32);
          Blo1 = load16(bl + (c + 1) * 32);
        }
        acc = __builtin_amdgcn_wmma_f32_16x16x32_bf16(false, Ahi[c], false, Bhi0,
                                                      (short)0, acc, false, false);
        acc = __builtin_amdgcn_wmma_f32_16x16x32_bf16(false, Alo[c], false, Bhi0,
                                                      (short)0, acc, false, false);
        acc = __builtin_amdgcn_wmma_f32_16x16x32_bf16(false, Ahi[c], false, Blo0,
                                                      (short)0, acc, false, false);
        Bhi0 = Bhi1;
        Blo0 = Blo1;
      }
      const float en = enorm[code];
      // score = ||e||^2 - 2 z.e  (||z||^2 is row-constant -> argmin-invariant)
#pragma unroll
      for (int r = 0; r < 8; ++r) {
        float s = fmaf(-2.0f, acc[r], en);
        if (s < best[r]) { best[r] = s; bidx[r] = code; }
      }
    }
    if (blk + 1 < NBLK) wait_async0();  // next slot's DMA fully landed
    __syncthreads();                    // all waves done reading / writing
  }

  // C layout: lane holds N = lane&15, rows M = (lane>>4)*8 + r.
  // Reduce the 16 lanes sharing each row; prefer lowest code on tie.
#pragma unroll
  for (int r = 0; r < 8; ++r) {
    float s = best[r];
    int   i = bidx[r];
#pragma unroll
    for (int off = 8; off >= 1; off >>= 1) {
      float s2 = __shfl_xor(s, off, 32);
      int   i2 = __shfl_xor(i, off, 32);
      if (s2 < s || (s2 == s && i2 < i)) { s = s2; i = i2; }
    }
    if (m == 0) {  // lanes 0 and 16 hold rows 0..7 and 8..15 of the tile
      int orow = blockIdx.x * MB + wave * 16 + hf * 8 + r;
      didx[orow] = i;
    }
  }
}

// ---------------------------------------------------------------------------
// Pass 3: gather z_q = emb[idx], write z_st = z + (z_q - z), idx as float,
// and a deterministic per-row partial of sum((z_q - z)^2) (no atomics).
// ---------------------------------------------------------------------------
__global__ __launch_bounds__(256) void vq_gather(const float* __restrict__ z,
                                                 const float* __restrict__ emb,
                                                 const int* __restrict__ didx,
                                                 float* __restrict__ out_zst,
                                                 float* __restrict__ out_idxf,
                                                 float* __restrict__ partial) {
  const int row = blockIdx.x, k = threadIdx.x;
  const int id = didx[row];
  float zq = emb[(size_t)id * H + k];
  float zv = z[(size_t)row * H + k];
  float d  = zq - zv;
  out_zst[(size_t)row * H + k] = zv + d;  // straight-through forward
  __shared__ float red[256];
  red[k] = d * d;
  __syncthreads();
  for (int s = 128; s > 0; s >>= 1) {
    if (k < s) red[k] += red[k + s];
    __syncthreads();
  }
  if (k == 0) {
    out_idxf[row] = (float)id;
    partial[row]  = red[0];
  }
}

// ---------------------------------------------------------------------------
// Pass 4: deterministic reduction of the 32768 partials -> both losses.
// ---------------------------------------------------------------------------
__global__ __launch_bounds__(256) void vq_final(const float* __restrict__ partial,
                                                float* __restrict__ out_loss) {
  const int tid = threadIdx.x;
  float s = 0.0f;
  for (int i = tid; i < NTOK; i += 256) s += partial[i];
  __shared__ float red[256];
  red[tid] = s;
  __syncthreads();
  for (int t = 128; t > 0; t >>= 1) {
    if (tid < t) red[tid] += red[tid + t];
    __syncthreads();
  }
  if (tid == 0) {
    float mval = red[0] / (float)((size_t)NTOK * H);
    out_loss[0] = mval;
    out_loss[1] = mval;
  }
}

extern "C" void kernel_launch(void* const* d_in, const int* in_sizes, int n_in,
                              void* d_out, int out_size, void* d_ws, size_t ws_size,
                              hipStream_t stream) {
  (void)in_sizes; (void)n_in; (void)out_size; (void)ws_size;
  const float* z   = (const float*)d_in[0];
  const float* emb = (const float*)d_in[1];

  float* out       = (float*)d_out;
  float* out_zst   = out;                       // 8*4096*256 floats
  float* out_idxf  = out + (size_t)NTOK * H;    // 32768 floats
  float* out_loss  = out_idxf + NTOK;           // 2 floats

  // Workspace layout (~8.7 MiB total).
  bf16_t* ehi  = (bf16_t*)d_ws;                  // 4 MiB
  bf16_t* elo  = ehi + (size_t)KCB * H;          // 4 MiB
  float*  enrm = (float*)(elo + (size_t)KCB * H);// 32 KiB
  int*    didx = (int*)(enrm + KCB);             // 128 KiB
  float*  part = (float*)(didx + NTOK);          // 128 KiB

  vq_prep  <<<KCB,       256, 0,    stream>>>(emb, ehi, elo, enrm);
  vq_argmin<<<NTOK / MB, 256, LDSB, stream>>>(z, ehi, elo, enrm, didx);
  vq_gather<<<NTOK,      256, 0,    stream>>>(z, emb, didx, out_zst, out_idxf, part);
  vq_final <<<1,         256, 0,    stream>>>(part, out_loss);
}
